// Attention_6219112645361
// MI455X (gfx1250) — matrix operand order repaired and checked by reading, not compiled, not run on hardware
//
#include <hip/hip_runtime.h>

// ---------------------------------------------------------------- types
typedef __bf16 bf16;
typedef bf16  v16bf __attribute__((ext_vector_type(16)));
typedef bf16  v8bf  __attribute__((ext_vector_type(8)));
typedef float v8f   __attribute__((ext_vector_type(8)));

static constexpr int Bd = 4, Td = 2048, Cd = 1024, Hd = 16, DHd = 64;
static constexpr int NQKV = 3 * Cd;        // 3072
static constexpr int MTOK = Bd * Td;       // 8192

union AFrag { v16bf v; v8bf h[2]; };

__device__ __forceinline__ v8f wmma_bf16(v16bf a, v16bf b, v8f c) {
  // D = A(16x32) x B(32x16) + C, fp32 accumulate
  return __builtin_amdgcn_wmma_f32_16x16x32_bf16(
      /*neg_a=*/false, a, /*neg_b=*/false, b,
      /*c_mod=*/(short)0, c, /*reuse_a=*/false, /*reuse_b=*/false);
}

// ---------------------------------------------------------------- fp32 -> bf16
__global__ void __launch_bounds__(256) cvt_bf16_kernel(
    const float* __restrict__ in, bf16* __restrict__ out, int n) {
  int i = blockIdx.x * 256 + threadIdx.x;
  if (i < n) out[i] = (bf16)in[i];
}

// ---------------------------------------------------------------- 64x64 GEMM core
// Per-wave 64(M) x 64(N) tile, K-contiguous row-major operands.
// 16 WMMAs per 8 fragment loads per K=32 chunk -> 32 FLOP/B from L2.
__device__ __forceinline__ void gemm_tile64(
    const bf16* __restrict__ A, const bf16* __restrict__ Bm,
    int m0, int n0, int K, int lo, int hi, v8f acc[4][4]) {
  const bf16* arow[4];
#pragma unroll
  for (int mt = 0; mt < 4; ++mt)
    arow[mt] = A + (size_t)(m0 + mt * 16 + lo) * K;

  for (int k = 0; k < K; k += 32) {
    AFrag af[4];
#pragma unroll
    for (int mt = 0; mt < 4; ++mt) {
      af[mt].h[0] = *(const v8bf*)(arow[mt] + k + 16 * hi);
      af[mt].h[1] = *(const v8bf*)(arow[mt] + k + 16 * hi + 8);
    }
#pragma unroll
    for (int nt = 0; nt < 4; ++nt) {
      v16bf bfr = *(const v16bf*)(Bm + (size_t)(n0 + nt * 16 + lo) * K + k + 16 * hi);
#pragma unroll
      for (int mt = 0; mt < 4; ++mt)
        acc[mt][nt] = wmma_bf16(af[mt].v, bfr, acc[mt][nt]);
    }
  }
}

// ---------------------------------------------------------------- QKV GEMM
// qkv = x @ W^T + b ; emit Q (pre-scaled by DH^-0.5) and K as [B,H,T,DH],
// V transposed as [B,H,DH,T].
__global__ void __launch_bounds__(128) qkv_gemm_kernel(
    const bf16* __restrict__ Xb, const bf16* __restrict__ Wb,
    const float* __restrict__ bias,
    bf16* __restrict__ Qo, bf16* __restrict__ Ko, bf16* __restrict__ Vt) {
  const int lane = threadIdx.x & 31;
  const int wave = threadIdx.x >> 5;
  const int lo = lane & 15, hi = lane >> 4;
  const int m0 = blockIdx.y * 64;
  const int n0 = blockIdx.x * 256 + wave * 64;

  const v8f vz = {0.f,0.f,0.f,0.f,0.f,0.f,0.f,0.f};
  v8f acc[4][4] = {{vz,vz,vz,vz},{vz,vz,vz,vz},{vz,vz,vz,vz},{vz,vz,vz,vz}};

  gemm_tile64(Xb, Wb, m0, n0, Cd, lo, hi, acc);

#pragma unroll
  for (int mt = 0; mt < 4; ++mt) {
#pragma unroll
    for (int nt = 0; nt < 4; ++nt) {
#pragma unroll
      for (int r = 0; r < 8; ++r) {
        const int n = n0 + nt * 16 + lo;            // 0..3071
        const int m = m0 + mt * 16 + 8 * hi + r;    // token
        float v = acc[mt][nt][r] + bias[n];
        const int which = n >> 10;                  // 0=q 1=k 2=v
        const int c = n & 1023;
        const int h = c >> 6, d = c & 63;
        const int b = m >> 11, t = m & 2047;
        const size_t bh = (size_t)(b * Hd + h);
        if (which == 0) {
          Qo[(bh * Td + t) * DHd + d] = (bf16)(v * 0.125f);   // DH^-0.5 = 1/8
        } else if (which == 1) {
          Ko[(bh * Td + t) * DHd + d] = (bf16)v;
        } else {
          Vt[(bh * DHd + d) * Td + t] = (bf16)v;              // transposed
        }
      }
    }
  }
}

// ---------------------------------------------------------------- flash attention
// One wave = 32 queries (two 16-row tiles) of one (b,h); online softmax over
// 32-key blocks.  K/V fragments are reused across both query tiles.
__global__ void __launch_bounds__(128) attn_kernel(
    const bf16* __restrict__ Q, const bf16* __restrict__ Km,
    const bf16* __restrict__ Vt, bf16* __restrict__ O) {
  __shared__ bf16 pbuf[4][32 * 32];     // per-wave P scratch (C-layout -> A-layout)

  const int lane = threadIdx.x & 31;
  const int wave = threadIdx.x >> 5;
  const int lo = lane & 15, hi = lane >> 4;
  const int bh = blockIdx.y;                       // b*H + h
  const int q0 = (blockIdx.x * 4 + wave) * 32;

  // Q fragments: [mtile][kchunk], resident in registers (pre-scaled by 1/8)
  AFrag qf[2][2];
#pragma unroll
  for (int mt = 0; mt < 2; ++mt) {
    const bf16* qrow = Q + ((size_t)bh * Td + q0 + mt * 16 + lo) * DHd;
#pragma unroll
    for (int kc = 0; kc < 2; ++kc) {
      qf[mt][kc].h[0] = *(const v8bf*)(qrow + kc * 32 + 16 * hi);
      qf[mt][kc].h[1] = *(const v8bf*)(qrow + kc * 32 + 16 * hi + 8);
    }
  }

  const v8f vz = {0.f,0.f,0.f,0.f,0.f,0.f,0.f,0.f};
  v8f accv[2][4] = {{vz,vz,vz,vz},{vz,vz,vz,vz}};
  float mrun[2][8], lrun[2][8];
#pragma unroll
  for (int mt = 0; mt < 2; ++mt)
#pragma unroll
    for (int r = 0; r < 8; ++r) { mrun[mt][r] = -3.0e38f; lrun[mt][r] = 0.f; }

  bf16* pw = &pbuf[wave][0];

  for (int kb = 0; kb < q0 + 32; kb += 32) {       // causal key blocks
    // K fragments for this block: [ntile][kchunk], shared by both q-tiles
    v16bf kf[2][2];
#pragma unroll
    for (int nt = 0; nt < 2; ++nt) {
      const bf16* krow = Km + ((size_t)bh * Td + kb + nt * 16 + lo) * DHd;
#pragma unroll
      for (int kc = 0; kc < 2; ++kc)
        kf[nt][kc] = *(const v16bf*)(krow + kc * 32 + 16 * hi);
    }

    // S = Q K^T  (32 x 32: 2 mtiles x 2 ntiles)
    v8f S[2][2] = {{vz, vz}, {vz, vz}};
#pragma unroll
    for (int mt = 0; mt < 2; ++mt)
#pragma unroll
      for (int nt = 0; nt < 2; ++nt)
#pragma unroll
        for (int kc = 0; kc < 2; ++kc)
          S[mt][nt] = wmma_bf16(qf[mt][kc].v, kf[nt][kc], S[mt][nt]);

#pragma unroll
    for (int mt = 0; mt < 2; ++mt) {
      // causal mask (wave-uniform predicate per q-tile)
      if (kb + 31 > q0 + mt * 16) {
#pragma unroll
        for (int r = 0; r < 8; ++r) {
          const int q = q0 + mt * 16 + 8 * hi + r;
          if (kb + lo > q)      S[mt][0][r] = -3.0e38f;
          if (kb + 16 + lo > q) S[mt][1][r] = -3.0e38f;
        }
      }

      // online softmax; row stats replicated across each 16-lane half
#pragma unroll
      for (int r = 0; r < 8; ++r) {
        float mx = fmaxf(S[mt][0][r], S[mt][1][r]);
        mx = fmaxf(mx, __shfl_xor(mx, 1, 16));
        mx = fmaxf(mx, __shfl_xor(mx, 2, 16));
        mx = fmaxf(mx, __shfl_xor(mx, 4, 16));
        mx = fmaxf(mx, __shfl_xor(mx, 8, 16));
        const float mn = fmaxf(mrun[mt][r], mx);
        const float alpha = __expf(mrun[mt][r] - mn);
        const float e0 = __expf(S[mt][0][r] - mn);
        const float e1 = __expf(S[mt][1][r] - mn);
        float rs = e0 + e1;
        rs += __shfl_xor(rs, 1, 16);
        rs += __shfl_xor(rs, 2, 16);
        rs += __shfl_xor(rs, 4, 16);
        rs += __shfl_xor(rs, 8, 16);
        lrun[mt][r] = lrun[mt][r] * alpha + rs;
        mrun[mt][r] = mn;
        S[mt][0][r] = e0; S[mt][1][r] = e1;
        accv[mt][0][r] *= alpha; accv[mt][1][r] *= alpha;
        accv[mt][2][r] *= alpha; accv[mt][3][r] *= alpha;
      }

      // C-layout -> A-layout staging of P through per-wave LDS
#pragma unroll
      for (int r = 0; r < 8; ++r) {
        pw[(mt * 16 + 8 * hi + r) * 32 + lo]      = (bf16)S[mt][0][r];
        pw[(mt * 16 + 8 * hi + r) * 32 + 16 + lo] = (bf16)S[mt][1][r];
      }
    }
    asm volatile("s_wait_dscnt 0x0" ::: "memory");

    AFrag pf[2];
#pragma unroll
    for (int mt = 0; mt < 2; ++mt) {
      pf[mt].h[0] = *(const v8bf*)(pw + (mt * 16 + lo) * 32 + 16 * hi);
      pf[mt].h[1] = *(const v8bf*)(pw + (mt * 16 + lo) * 32 + 16 * hi + 8);
    }
    asm volatile("" ::: "memory");

    // O += P @ V   (V stored transposed -> B-frag is K-contiguous)
#pragma unroll
    for (int nt = 0; nt < 4; ++nt) {
      v16bf vf = *(const v16bf*)(Vt + ((size_t)bh * DHd + nt * 16 + lo) * Td + kb + 16 * hi);
#pragma unroll
      for (int mt = 0; mt < 2; ++mt)
        accv[mt][nt] = wmma_bf16(pf[mt].v, vf, accv[mt][nt]);
    }
  }

  // epilogue: normalize, write [B,T,C] bf16
  const int b = bh >> 4, h = bh & 15;
#pragma unroll
  for (int mt = 0; mt < 2; ++mt) {
#pragma unroll
    for (int r = 0; r < 8; ++r) {
      const float inv = 1.0f / lrun[mt][r];
      const int q = q0 + mt * 16 + 8 * hi + r;
      bf16* orow = O + ((size_t)(b * Td + q)) * Cd + h * DHd;
#pragma unroll
      for (int nt = 0; nt < 4; ++nt)
        orow[nt * 16 + lo] = (bf16)(accv[mt][nt][r] * inv);
    }
  }
}

// ---------------------------------------------------------------- output projection
__global__ void __launch_bounds__(128) proj_gemm_kernel(
    const bf16* __restrict__ Ab, const bf16* __restrict__ Wb,
    const float* __restrict__ bias, float* __restrict__ out) {
  const int lane = threadIdx.x & 31;
  const int wave = threadIdx.x >> 5;
  const int lo = lane & 15, hi = lane >> 4;
  const int m0 = blockIdx.y * 64;
  const int n0 = blockIdx.x * 256 + wave * 64;

  const v8f vz = {0.f,0.f,0.f,0.f,0.f,0.f,0.f,0.f};
  v8f acc[4][4] = {{vz,vz,vz,vz},{vz,vz,vz,vz},{vz,vz,vz,vz},{vz,vz,vz,vz}};

  gemm_tile64(Ab, Wb, m0, n0, Cd, lo, hi, acc);

#pragma unroll
  for (int mt = 0; mt < 4; ++mt) {
#pragma unroll
    for (int nt = 0; nt < 4; ++nt) {
#pragma unroll
      for (int r = 0; r < 8; ++r) {
        const int n = n0 + nt * 16 + lo;
        const int m = m0 + mt * 16 + 8 * hi + r;
        out[(size_t)m * Cd + n] = acc[mt][nt][r] + bias[n];
      }
    }
  }
}

// ---------------------------------------------------------------- launch
extern "C" void kernel_launch(void* const* d_in, const int* in_sizes, int n_in,
                              void* d_out, int out_size, void* d_ws, size_t ws_size,
                              hipStream_t stream) {
  const float* x      = (const float*)d_in[0];
  const float* qkv_w  = (const float*)d_in[1];
  const float* qkv_b  = (const float*)d_in[2];
  const float* proj_w = (const float*)d_in[3];
  const float* proj_b = (const float*)d_in[4];
  // d_in[5] = causal mask: handled analytically, not read.
  float* out = (float*)d_out;

  char* ws = (char*)d_ws;
  size_t off = 0;
  bf16* x_bf  = (bf16*)(ws + off); off += (size_t)MTOK * Cd * 2;   // 16 MB
  bf16* wqkv  = (bf16*)(ws + off); off += (size_t)NQKV * Cd * 2;   //  6 MB
  bf16* wproj = (bf16*)(ws + off); off += (size_t)Cd * Cd * 2;     //  2 MB
  bf16* Qb    = (bf16*)(ws + off); off += (size_t)MTOK * Cd * 2;   // 16 MB  [B,H,T,DH]
  bf16* Kb    = (bf16*)(ws + off); off += (size_t)MTOK * Cd * 2;   // 16 MB  [B,H,T,DH]
  bf16* Vt    = (bf16*)(ws + off); off += (size_t)MTOK * Cd * 2;   // 16 MB  [B,H,DH,T]
  bf16* aout  = (bf16*)(ws + off); off += (size_t)MTOK * Cd * 2;   // 16 MB  [B,T,C]
  (void)ws_size; (void)in_sizes; (void)n_in; (void)out_size;

  const int n_x = MTOK * Cd, n_wq = NQKV * Cd, n_wp = Cd * Cd;
  cvt_bf16_kernel<<<(n_x  + 255) / 256, 256, 0, stream>>>(x,      x_bf,  n_x);
  cvt_bf16_kernel<<<(n_wq + 255) / 256, 256, 0, stream>>>(qkv_w,  wqkv,  n_wq);
  cvt_bf16_kernel<<<(n_wp + 255) / 256, 256, 0, stream>>>(proj_w, wproj, n_wp);

  // QKV:  M=8192, N=3072 -> grid (3072/256, 8192/64), 64x64 tile per wave
  qkv_gemm_kernel<<<dim3(NQKV / 256, MTOK / 64), 128, 0, stream>>>(
      x_bf, wqkv, qkv_b, Qb, Kb, Vt);

  // attention: 64 q-tiles (32 queries each) per (b,h), 4 waves/block
  attn_kernel<<<dim3((Td / 32) / 4, Bd * Hd), 128, 0, stream>>>(Qb, Kb, Vt, aout);

  // projection: M=8192, N=1024
  proj_gemm_kernel<<<dim3(Cd / 256, MTOK / 64), 128, 0, stream>>>(
      aout, wproj, proj_b, out);
}
